// Uniformity_49589692399687
// MI455X (gfx1250) — compile-verified, matrix operand order checked by reading, not run on hardware
//
#include <hip/hip_runtime.h>
#include <math.h>

// Problem shape (fixed by the reference).
#define N_ROWS 2048
#define DIM    512
#define QN     3
#define KN     8192
#define QK     (QN * KN)

typedef __bf16 bf16;
typedef __attribute__((ext_vector_type(16))) __bf16 v16bf;
typedef __attribute__((ext_vector_type(8)))  float  v8f;

union FragCast {
  uint4 u[2];
  v16bf v;
};

// Load one 16-element bf16 WMMA fragment for this lane.
// ISA 16-bit A layout (16x32): lanes 0-15 hold K = [0..7] (VGPR0-3) and
// [16..23] (VGPR4-7); lanes 16-31 hold K = [8..15] and [24..31].
// => per lane: two contiguous 16-byte loads at elem offsets koff and koff+16.
__device__ __forceinline__ v16bf load_frag(const bf16* row_k, int koff) {
  FragCast f;
  f.u[0] = *reinterpret_cast<const uint4*>(row_k + koff);
  f.u[1] = *reinterpret_cast<const uint4*>(row_k + koff + 16);
  return f.v;
}

// ---------------------------------------------------------------------------
// Kernel 1: per-row L2 normalize, emit bf16 hi/lo split planes.
// One 128-thread block (4 waves) per row.
// ---------------------------------------------------------------------------
__global__ void __launch_bounds__(128)
k_rownorm(const float* __restrict__ src, bf16* __restrict__ hi,
          bf16* __restrict__ lo, int d) {
  const int row = blockIdx.x;
  const float* r = src + (size_t)row * d;
  float ss = 0.f;
  for (int i = threadIdx.x; i < d; i += blockDim.x) {
    float v = r[i];
    ss = fmaf(v, v, ss);
  }
  ss += __shfl_xor(ss, 1);
  ss += __shfl_xor(ss, 2);
  ss += __shfl_xor(ss, 4);
  ss += __shfl_xor(ss, 8);
  ss += __shfl_xor(ss, 16);
  __shared__ float warp_s[4];
  if ((threadIdx.x & 31) == 0) warp_s[threadIdx.x >> 5] = ss;
  __syncthreads();
  const float tot = warp_s[0] + warp_s[1] + warp_s[2] + warp_s[3];
  const float inv = 1.0f / fmaxf(sqrtf(tot), 1e-12f);
  for (int i = threadIdx.x; i < d; i += blockDim.x) {
    const float v = r[i] * inv;
    const bf16 h = (bf16)v;                 // round-to-nearest-even
    hi[(size_t)row * d + i] = h;
    lo[(size_t)row * d + i] = (bf16)(v - (float)h);  // residual
  }
}

// ---------------------------------------------------------------------------
// Kernel 2: zero the per-row accumulators and the output scalar.
// ---------------------------------------------------------------------------
__global__ void k_zero(float* __restrict__ total_acc, float* __restrict__ masked_acc,
                       float* __restrict__ out, int n) {
  const int i = blockIdx.x * blockDim.x + threadIdx.x;
  if (i < n) {
    total_acc[i]  = 0.f;
    masked_acc[i] = 0.f;
  }
  if (i == 0) out[0] = 0.f;
}

// ---------------------------------------------------------------------------
// Kernel 3: tiled similarity + exp + row reduction.
// Block = 256 threads = 8 waves arranged 2 (M) x 4 (N): block tile 32x256.
// Each wave owns a 16x64 strip = four 16x16 accumulators sharing one pair of
// A fragments per k-step (12 WMMAs per 20 b128 loads).
// Split-bf16 product: sim ~= ah*bh + ah*bl + al*bh (f32 WMMA accumulate).
// ---------------------------------------------------------------------------
__global__ void __launch_bounds__(256)
k_sim_accum(const bf16* __restrict__ xhi, const bf16* __restrict__ xlo,
            const bf16* __restrict__ qhi, const bf16* __restrict__ qlo,
            const int* __restrict__ targets, const int* __restrict__ q_targets,
            const int* __restrict__ d_order,
            float* __restrict__ total_acc, float* __restrict__ masked_acc) {
  const int wave   = threadIdx.x >> 5;
  const int lane   = threadIdx.x & 31;
  const int wave_m = wave >> 2;              // 0..1
  const int wave_n = wave & 3;               // 0..3
  const int mblk   = blockIdx.y * 32;        // block row base (32 rows)
  const int m0     = mblk + wave_m * 16;     // this wave's row tile base
  const int cblk   = blockIdx.x * 256;       // block column base (256 cols)
  const int c0     = cblk + wave_n * 64;     // this wave's 64-col strip base
  const int order  = d_order[0];
  // 8192 % 256 == 0, so every block's 256 columns live in exactly one queue.
  const bool is_order_blk = (cblk / KN) == order;

  __shared__ float s_tot[32];
  __shared__ float s_msk[32];
  __shared__ int   s_targ[32];
  __shared__ int   s_qt[256];

  if (threadIdx.x < 32) {
    s_tot[threadIdx.x]  = 0.f;
    s_msk[threadIdx.x]  = 0.f;
    s_targ[threadIdx.x] = targets[mblk + threadIdx.x];
  }
  if (is_order_blk) {
    s_qt[threadIdx.x] = q_targets[cblk + threadIdx.x];  // flat == order*KN + j
  }
  __syncthreads();

  const int m    = lane & 15;
  const int koff = (lane >> 4) * 8;
  const bf16* arow_hi = xhi + (size_t)(m0 + m) * DIM;
  const bf16* arow_lo = xlo + (size_t)(m0 + m) * DIM;
  const bf16* brow_hi = qhi + (size_t)(c0 + m) * DIM;
  const bf16* brow_lo = qlo + (size_t)(c0 + m) * DIM;

  v8f acc[4];
  acc[0] = v8f{};
  acc[1] = v8f{};
  acc[2] = v8f{};
  acc[3] = v8f{};

  for (int kk = 0; kk < DIM; kk += 32) {
    const v16bf ah = load_frag(arow_hi + kk, koff);
    const v16bf al = load_frag(arow_lo + kk, koff);
#pragma unroll
    for (int s = 0; s < 4; ++s) {
      const size_t soff = (size_t)s * 16 * DIM;  // next 16-row B sub-tile
      const v16bf bh = load_frag(brow_hi + soff + kk, koff);
      const v16bf bl = load_frag(brow_lo + soff + kk, koff);
      acc[s] = __builtin_amdgcn_wmma_f32_16x16x32_bf16(false, ah, false, bh,
                                                       (short)0, acc[s], false, false);
      acc[s] = __builtin_amdgcn_wmma_f32_16x16x32_bf16(false, ah, false, bl,
                                                       (short)0, acc[s], false, false);
      acc[s] = __builtin_amdgcn_wmma_f32_16x16x32_bf16(false, al, false, bh,
                                                       (short)0, acc[s], false, false);
    }
  }

  // C/D layout: VGPR r, lane L -> row (r + 8*(L>>4)), col (L&15).
  const int half     = lane >> 4;
  const int col_lane = lane & 15;
  int qts[4] = {0, 0, 0, 0};
  if (is_order_blk) {
#pragma unroll
    for (int s = 0; s < 4; ++s) qts[s] = s_qt[wave_n * 64 + s * 16 + col_lane];
  }

#pragma unroll
  for (int r = 0; r < 8; ++r) {
    const int row_r = wave_m * 16 + half * 8 + r;  // row within 32-row block
    float es[4];
#pragma unroll
    for (int s = 0; s < 4; ++s)
      es[s] = __expf(fmaf(4.0f, acc[s][r], -4.0f));  // exp(-T*(2-2*sim)), T=2

    float rs = (es[0] + es[1]) + (es[2] + es[3]);    // 64 columns -> 16 lanes
    rs += __shfl_xor(rs, 1);
    rs += __shfl_xor(rs, 2);
    rs += __shfl_xor(rs, 4);
    rs += __shfl_xor(rs, 8);
    if (col_lane == 0) atomicAdd(&s_tot[row_r], rs);  // ds_add_f32

    if (is_order_blk) {
      const int t = s_targ[row_r];
      float me = 0.f;
#pragma unroll
      for (int s = 0; s < 4; ++s) me += (t == qts[s]) ? es[s] : 0.f;
      me += __shfl_xor(me, 1);
      me += __shfl_xor(me, 2);
      me += __shfl_xor(me, 4);
      me += __shfl_xor(me, 8);
      if (col_lane == 0) atomicAdd(&s_msk[row_r], me);
    }
  }
  __syncthreads();

  if (threadIdx.x < 32) {
    unsafeAtomicAdd(&total_acc[mblk + threadIdx.x], s_tot[threadIdx.x]);
    if (is_order_blk)
      unsafeAtomicAdd(&masked_acc[mblk + threadIdx.x], s_msk[threadIdx.x]);
  }
}

// ---------------------------------------------------------------------------
// Kernel 4: per-row count + log term, mean into d_out. One block per row.
// ---------------------------------------------------------------------------
__global__ void __launch_bounds__(256)
k_finalize(const float* __restrict__ total_acc, const float* __restrict__ masked_acc,
           const int* __restrict__ targets, const int* __restrict__ q_targets,
           const int* __restrict__ d_order, float* __restrict__ out) {
  const int row = blockIdx.x;
  const int order = d_order[0];
  const int t = targets[row];
  const int* qt = q_targets + (size_t)order * KN;
  int cnt = 0;
  for (int j = threadIdx.x; j < KN; j += blockDim.x) cnt += (qt[j] == t) ? 1 : 0;
  cnt += __shfl_xor(cnt, 1);
  cnt += __shfl_xor(cnt, 2);
  cnt += __shfl_xor(cnt, 4);
  cnt += __shfl_xor(cnt, 8);
  cnt += __shfl_xor(cnt, 16);
  __shared__ int s_cnt[8];
  if ((threadIdx.x & 31) == 0) s_cnt[threadIdx.x >> 5] = cnt;
  __syncthreads();
  if (threadIdx.x == 0) {
    int c = 0;
    for (int w = 0; w < 8; ++w) c += s_cnt[w];
    const float count = (float)(QN * KN - c);
    const float val = logf((total_acc[row] - masked_acc[row]) / count);
    unsafeAtomicAdd(out, val * (1.0f / (float)N_ROWS));
  }
}

// ---------------------------------------------------------------------------
// Launch
// ---------------------------------------------------------------------------
extern "C" void kernel_launch(void* const* d_in, const int* in_sizes, int n_in,
                              void* d_out, int out_size, void* d_ws, size_t ws_size,
                              hipStream_t stream) {
  const float* x         = (const float*)d_in[0];
  const float* q_data    = (const float*)d_in[1];
  const int*   targets   = (const int*)d_in[2];
  const int*   q_targets = (const int*)d_in[3];
  const int*   order     = (const int*)d_in[4];
  float* out = (float*)d_out;

  char* ws = (char*)d_ws;
  size_t off = 0;
  auto alloc = [&](size_t bytes) -> void* {
    void* p = ws + off;
    off = (off + bytes + 255) & ~(size_t)255;
    return p;
  };
  bf16* xhi = (bf16*)alloc((size_t)N_ROWS * DIM * sizeof(bf16));
  bf16* xlo = (bf16*)alloc((size_t)N_ROWS * DIM * sizeof(bf16));
  bf16* qhi = (bf16*)alloc((size_t)QK * DIM * sizeof(bf16));
  bf16* qlo = (bf16*)alloc((size_t)QK * DIM * sizeof(bf16));
  float* total_acc  = (float*)alloc((size_t)N_ROWS * sizeof(float));
  float* masked_acc = (float*)alloc((size_t)N_ROWS * sizeof(float));

  // 1) normalize rows -> bf16 hi/lo planes
  k_rownorm<<<N_ROWS, 128, 0, stream>>>(x, xhi, xlo, DIM);
  k_rownorm<<<QK, 128, 0, stream>>>(q_data, qhi, qlo, DIM);

  // 2) zero accumulators + output (must happen every call)
  k_zero<<<(N_ROWS + 255) / 256, 256, 0, stream>>>(total_acc, masked_acc, out, N_ROWS);

  // 3) WMMA similarity + exp + row sums (block tile 32x256, wave tile 16x64)
  dim3 grid(QK / 256, N_ROWS / 32);
  k_sim_accum<<<grid, 256, 0, stream>>>(xhi, xlo, qhi, qlo, targets, q_targets,
                                        order, total_acc, masked_acc);

  // 4) counts + log-mean
  k_finalize<<<N_ROWS, 256, 0, stream>>>(total_acc, masked_acc, targets,
                                         q_targets, order, out);
}
